// RGCN_84894323573129
// MI455X (gfx1250) — matrix operand order repaired and checked by reading, not compiled, not run on hardware
//
#include <hip/hip_runtime.h>

#define NP 60000
#define ND 8000
#define NE 600000
#define DIN 768
#define DHID 256

typedef __attribute__((ext_vector_type(16))) __bf16 v16bf;
typedef __attribute__((ext_vector_type(8)))  __bf16 v8bf;
typedef __attribute__((ext_vector_type(8)))  float  v8f;

union ABf { v16bf v; v8bf h[2]; };

static __device__ __forceinline__ unsigned short f2bf(float f) {
  unsigned int u = __float_as_uint(f);
  u += 0x7FFFu + ((u >> 16) & 1u);           // round-to-nearest-even
  return (unsigned short)(u >> 16);
}

static __device__ __forceinline__ v8f vzero8() {
  v8f z;
#pragma unroll
  for (int i = 0; i < 8; ++i) z[i] = 0.0f;
  return z;
}

// CDNA5 async global->LDS copies (ASYNCcnt-tracked DMA; inline asm is the
// toolchain-portable spelling per the bridge doc).
static __device__ __forceinline__ void async_cp16(unsigned lds_off,
                                                  unsigned long long gaddr) {
  asm volatile("global_load_async_to_lds_b128 %0, %1, off"
               :: "v"(lds_off), "v"(gaddr) : "memory");
}
static __device__ __forceinline__ void async_cp8(unsigned lds_off,
                                                 unsigned long long gaddr) {
  asm volatile("global_load_async_to_lds_b64 %0, %1, off"
               :: "v"(lds_off), "v"(gaddr) : "memory");
}

// ---------------------------------------------------------------- utilities
__global__ __launch_bounds__(256) void k_zero4(float4* __restrict__ p, long long n4) {
  long long i = (long long)blockIdx.x * blockDim.x + threadIdx.x;
  if (i < n4) p[i] = make_float4(0.f, 0.f, 0.f, 0.f);
}

__global__ __launch_bounds__(256) void k_cast_bf16(const float* __restrict__ x,
                                                   unsigned short* __restrict__ y,
                                                   long long n) {
  long long i = ((long long)blockIdx.x * blockDim.x + threadIdx.x) * 4;
  if (i >= n) return;
  float4 v = *(const float4*)(x + i);
  ushort4 r = make_ushort4(f2bf(v.x), f2bf(v.y), f2bf(v.z), f2bf(v.w));
  *(ushort4*)(y + i) = r;
}

// Wt[n*K + k] = bf16(W[k*N + n])   (weights are small; done once per launch)
__global__ __launch_bounds__(256) void k_transpose_cast(const float* __restrict__ W,
                                                        unsigned short* __restrict__ Wt,
                                                        int K, int N) {
  long long t = (long long)blockIdx.x * blockDim.x + threadIdx.x;
  if (t >= (long long)K * N) return;
  int k = (int)(t / N), n = (int)(t % N);
  Wt[(size_t)n * K + k] = f2bf(W[(size_t)k * N + n]);
}

// ---------------------------------------------------------------- WMMA GEMM
// C[M,256] = A[M,K](bf16 row-major) @ B[K,256], B supplied transposed as
// Bt[256,K] bf16 row-major.  Optional addend (may alias C).
//
// Block: 256 thr = 8 waves, tile 32(M) x 256(N), K-step 32.
// LDS double buffer per K-step: A 32x32 (2KB) + Bt-slice 256x32 (16KB).
// Staging via async DMA: per wave per stage exactly 5 async instructions
// (1x b64 for A + 4x b128 for B) -> pipeline wait is s_wait_asynccnt 5.
#define GEMM_BUF_USH 9216   // (2048 + 16384) bytes / 2
__global__ __launch_bounds__(256) void k_gemm_bf16_n256(const __bf16* __restrict__ A,
                                                        const __bf16* __restrict__ Bt,
                                                        const float* __restrict__ Add,
                                                        float* __restrict__ C,
                                                        int M, int K) {
  __shared__ __align__(16) unsigned short smem[2 * GEMM_BUF_USH];

  const int tid  = threadIdx.x;
  const int lane = tid & 31;
  const int wave = tid >> 5;
  const int mBlock = blockIdx.x * 32;
  const int mTile  = (wave >> 2) * 16;      // 0 or 16
  const int nBase  = (wave & 3) * 64;       // 0/64/128/192
  const bool hi = lane >= 16;
  const int  l15 = lane & 15;

  // LDS byte address of smem base (low 32 bits of the generic pointer are the
  // wave-relative LDS byte address on CDNA5).
  const unsigned ldsBase = (unsigned)(unsigned long long)(const void*)&smem[0];

  // Per-thread staging assignments.
  // A: 32 rows x 64B -> 256 x 8B chunks; chunk c = tid: row tid>>3, off tid&7.
  const unsigned long long gA =
      (unsigned long long)((const char*)(A + (size_t)(mBlock + (tid >> 3)) * K) +
                           (size_t)(tid & 7) * 8);
  // B: 256 rows x 64B -> 1024 x 16B chunks; chunk c = tid + 256*i.
  const unsigned long long gBt = (unsigned long long)(const char*)Bt;

  v8f acc[4];
#pragma unroll
  for (int j = 0; j < 4; ++j) acc[j] = vzero8();

  const int nk = K >> 5;                    // K / 32

  auto stage = [&](int buf, int k0) {
    const unsigned bufBase = ldsBase + (unsigned)buf * (GEMM_BUF_USH * 2);
    async_cp8(bufBase + (unsigned)tid * 8, gA + (unsigned long long)k0 * 2ull);
#pragma unroll
    for (int i = 0; i < 4; ++i) {
      const int c = tid + 256 * i;
      const int row = c >> 2, off = c & 3;  // 4 x 16B chunks per 64B row
      const unsigned long long g =
          gBt + (unsigned long long)row * (unsigned long long)K * 2ull +
          (unsigned long long)k0 * 2ull + (unsigned long long)off * 16ull;
      async_cp16(bufBase + 2048u + (unsigned)c * 16u, g);
    }
  };

  stage(0, 0);

  for (int kt = 0; kt < nk; ++kt) {
    const int cur = kt & 1;
    if (kt + 1 < nk) {
      stage(cur ^ 1, (kt + 1) << 5);
      asm volatile("s_wait_asynccnt 0x5" ::: "memory");   // current tile done
    } else {
      asm volatile("s_wait_asynccnt 0x0" ::: "memory");
    }
    __syncthreads();

    const unsigned short* sb = smem + (size_t)cur * GEMM_BUF_USH;
    const __bf16* sA = (const __bf16*)sb;                 // 32 rows x 32 halfs
    const __bf16* sB = (const __bf16*)(sb + 1024);        // 256 rows x 32 halfs

    // A fragment: row = mTile + l15;
    // lo lanes hold K {0..15}, hi lanes {16..31}.
    ABf a;
    const __bf16* aRow = sA + (size_t)(mTile + l15) * 32 + (hi ? 16 : 0);
    a.h[0] = *(const v8bf*)(aRow);
    a.h[1] = *(const v8bf*)(aRow + 8);

#pragma unroll
    for (int j = 0; j < 4; ++j) {
      // B fragment: col = nBase + j*16 + l15; K {0..15} (lo) / {16..31} (hi).
      const v16bf b = *(const v16bf*)(sB + (size_t)(nBase + j * 16 + l15) * 32 +
                                      (hi ? 16 : 0));
      acc[j] = __builtin_amdgcn_wmma_f32_16x16x32_bf16(
          false, a.v, false, b, (short)0, acc[j], false, false);
    }
    __syncthreads();   // protect this buffer from the next stage's DMA writes
  }

  // C/D layout: VGPR v -> row mBase + (hi?8:0) + v, col = nBase + j*16 + l15
  const int rowOff = (hi ? 8 : 0) + mBlock + mTile;
#pragma unroll
  for (int j = 0; j < 4; ++j) {
#pragma unroll
    for (int v = 0; v < 8; ++v) {
      size_t idx = (size_t)(rowOff + v) * DHID + nBase + j * 16 + l15;
      float val = acc[j][v];
      if (Add) val += Add[idx];
      C[idx] = val;
    }
  }
}

// ---------------------------------------------------------------- scatter
__global__ __launch_bounds__(256) void k_scatter_add(const float* __restrict__ msg,
                                                     const int* __restrict__ src,
                                                     const int* __restrict__ dst,
                                                     float* __restrict__ agg,
                                                     long long nE) {
  long long t = (long long)blockIdx.x * blockDim.x + threadIdx.x;
  long long e = t >> 6;                 // 64 threads per edge (4 cols each)
  if (e >= nE) return;
  int c = (int)(t & 63) << 2;
  int s = src[e], d = dst[e];
  float4 m = *(const float4*)(msg + (size_t)s * DHID + c);
  float* o = agg + (size_t)d * DHID + c;
  atomicAdd(o + 0, m.x);
  atomicAdd(o + 1, m.y);
  atomicAdd(o + 2, m.z);
  atomicAdd(o + 3, m.w);
}

__global__ __launch_bounds__(256) void k_count(const int* __restrict__ dst,
                                               float* __restrict__ cnt, long long nE) {
  long long e = (long long)blockIdx.x * blockDim.x + threadIdx.x;
  if (e < nE) atomicAdd(cnt + dst[e], 1.0f);
}

// ---------------------------------------------------------------- epilogue
// v = agg/max(cnt,1) + add1 + bias1 (+ add2)(+ bias2); relu; LayerNorm.
// One block (256 thr) per row; D == 256.
__global__ __launch_bounds__(256) void k_epilogue(const float* __restrict__ agg,
                                                  const float* __restrict__ cnt,
                                                  const float* __restrict__ add1,
                                                  const float* __restrict__ add2,
                                                  const float* __restrict__ bias1,
                                                  const float* __restrict__ bias2,
                                                  const float* __restrict__ gamma,
                                                  const float* __restrict__ beta,
                                                  float* __restrict__ outF,
                                                  unsigned short* __restrict__ outBF) {
  const int r = blockIdx.x;
  const int c = threadIdx.x;
  const size_t idx = (size_t)r * DHID + c;
  float inv = 1.0f / fmaxf(cnt[r], 1.0f);
  float v = agg[idx] * inv + add1[idx] + bias1[c];
  if (add2) v += add2[idx];
  if (bias2) v += bias2[c];
  v = fmaxf(v, 0.0f);

  __shared__ float sS[8], sQ[8];
  float s = v, q = v * v;
#pragma unroll
  for (int o = 16; o > 0; o >>= 1) {
    s += __shfl_down(s, o);
    q += __shfl_down(q, o);
  }
  const int lane = c & 31, w = c >> 5;
  if (lane == 0) { sS[w] = s; sQ[w] = q; }
  __syncthreads();
  if (w == 0) {
    float s2 = (lane < 8) ? sS[lane] : 0.0f;
    float q2 = (lane < 8) ? sQ[lane] : 0.0f;
#pragma unroll
    for (int o = 4; o > 0; o >>= 1) {
      s2 += __shfl_down(s2, o);
      q2 += __shfl_down(q2, o);
    }
    if (lane == 0) { sS[0] = s2; sQ[0] = q2; }
  }
  __syncthreads();
  const float mu = sS[0] * (1.0f / DHID);
  const float var = sQ[0] * (1.0f / DHID) - mu * mu;
  const float y = (v - mu) * rsqrtf(var + 1e-5f) * gamma[c] + beta[c];
  outF[idx] = y;
  if (outBF) outBF[idx] = f2bf(y);
}

// ---------------------------------------------------------------- driver
extern "C" void kernel_launch(void* const* d_in, const int* in_sizes, int n_in,
                              void* d_out, int out_size, void* d_ws, size_t ws_size,
                              hipStream_t stream) {
  const float* x_paper   = (const float*)d_in[0];
  const float* x_dataset = (const float*)d_in[1];
  const int*   src_use   = (const int*)d_in[2];
  const int*   dst_use   = (const int*)d_in[3];
  const int*   src_rev   = (const int*)d_in[4];
  const int*   dst_rev   = (const int*)d_in[5];
  const float* w1_use    = (const float*)d_in[6];
  const float* root1_use = (const float*)d_in[7];
  const float* b1_use    = (const float*)d_in[8];
  const float* w1_rev    = (const float*)d_in[9];
  const float* root1_rev = (const float*)d_in[10];
  const float* b1_rev    = (const float*)d_in[11];
  const float* res_w     = (const float*)d_in[12];
  const float* res_b     = (const float*)d_in[13];
  const float* ln1_g     = (const float*)d_in[14];
  const float* ln1_b     = (const float*)d_in[15];
  const float* w2_use    = (const float*)d_in[16];
  const float* root2_use = (const float*)d_in[17];
  const float* b2_use    = (const float*)d_in[18];
  const float* w2_rev    = (const float*)d_in[19];
  const float* root2_rev = (const float*)d_in[20];
  const float* b2_rev    = (const float*)d_in[21];
  const float* ln2_g     = (const float*)d_in[22];
  const float* ln2_b     = (const float*)d_in[23];

  float* y_pp = (float*)d_out;
  float* y_ds = y_pp + (size_t)NP * DHID;

  // ---- workspace carve-up
  size_t off = 0;
  auto alloc = [&](size_t bytes) -> void* {
    void* p = (char*)d_ws + off;
    off += (bytes + 255) & ~(size_t)255;
    return p;
  };
  unsigned short* xPbf = (unsigned short*)alloc((size_t)NP * DIN * 2);
  unsigned short* xDbf = (unsigned short*)alloc((size_t)ND * DIN * 2);
  unsigned short* w1u_t = (unsigned short*)alloc((size_t)DIN * DHID * 2);
  unsigned short* r1u_t = (unsigned short*)alloc((size_t)DIN * DHID * 2);
  unsigned short* w1r_t = (unsigned short*)alloc((size_t)DIN * DHID * 2);
  unsigned short* r1r_t = (unsigned short*)alloc((size_t)DIN * DHID * 2);
  unsigned short* res_t = (unsigned short*)alloc((size_t)DIN * DHID * 2);
  unsigned short* w2u_t = (unsigned short*)alloc((size_t)DHID * DHID * 2);
  unsigned short* r2u_t = (unsigned short*)alloc((size_t)DHID * DHID * 2);
  unsigned short* w2r_t = (unsigned short*)alloc((size_t)DHID * DHID * 2);
  unsigned short* r2r_t = (unsigned short*)alloc((size_t)DHID * DHID * 2);
  float* msgP  = (float*)alloc((size_t)NP * DHID * 4);
  float* msgD  = (float*)alloc((size_t)ND * DHID * 4);
  float* aggP  = (float*)alloc((size_t)NP * DHID * 4);
  float* aggD  = (float*)alloc((size_t)ND * DHID * 4);
  float* rootP = (float*)alloc((size_t)NP * DHID * 4);
  float* rootD = (float*)alloc((size_t)ND * DHID * 4);
  float* hP    = (float*)alloc((size_t)NP * DHID * 4);
  float* hD    = (float*)alloc((size_t)ND * DHID * 4);
  unsigned short* hPbf = (unsigned short*)alloc((size_t)NP * DHID * 2);
  unsigned short* hDbf = (unsigned short*)alloc((size_t)ND * DHID * 2);
  float* cntP = (float*)alloc((size_t)NP * 4);
  float* cntD = (float*)alloc((size_t)ND * 4);

  auto blk = [](long long n) { return (unsigned)((n + 255) / 256); };
  auto zero = [&](float* p, long long n) {
    k_zero4<<<blk(n / 4), 256, 0, stream>>>((float4*)p, n / 4);
  };
  auto cast = [&](const float* x, unsigned short* y, long long n) {
    k_cast_bf16<<<blk(n / 4), 256, 0, stream>>>(x, y, n);
  };
  auto tcast = [&](const float* W, unsigned short* Wt, int K) {
    k_transpose_cast<<<blk((long long)K * DHID), 256, 0, stream>>>(W, Wt, K, DHID);
  };
  auto gemm = [&](const unsigned short* A, const unsigned short* Bt,
                  const float* Add, float* C, int M, int K) {
    k_gemm_bf16_n256<<<dim3(M / 32), dim3(256), 0, stream>>>(
        (const __bf16*)A, (const __bf16*)Bt, Add, C, M, K);
  };
  auto scatter = [&](const float* msg, const int* s, const int* d, float* agg) {
    k_scatter_add<<<blk((long long)NE * 64), 256, 0, stream>>>(msg, s, d, agg, NE);
  };

  // ---- precompute: bf16 activations + transposed bf16 weights
  cast(x_paper, xPbf, (long long)NP * DIN);
  cast(x_dataset, xDbf, (long long)ND * DIN);
  tcast(w1_use, w1u_t, DIN);   tcast(root1_use, r1u_t, DIN);
  tcast(w1_rev, w1r_t, DIN);   tcast(root1_rev, r1r_t, DIN);
  tcast(res_w, res_t, DIN);
  tcast(w2_use, w2u_t, DHID);  tcast(root2_use, r2u_t, DHID);
  tcast(w2_rev, w2r_t, DHID);  tcast(root2_rev, r2r_t, DHID);

  // ---- counts (shared by both layers)
  zero(cntP, NP); zero(cntD, ND);
  k_count<<<blk(NE), 256, 0, stream>>>(dst_use, cntD, NE);
  k_count<<<blk(NE), 256, 0, stream>>>(dst_rev, cntP, NE);

  // ---- layer 1 GEMMs
  gemm(xPbf, w1u_t, nullptr, msgP, NP, DIN);            // msgs paper->ds
  gemm(xDbf, w1r_t, nullptr, msgD, ND, DIN);            // msgs ds->paper
  gemm(xDbf, r1u_t, nullptr, rootD, ND, DIN);           // root term (ds)
  gemm(xDbf, res_t, rootD, rootD, ND, DIN);             // += residual proj
  gemm(xPbf, r1r_t, nullptr, rootP, NP, DIN);           // root term (pp)
  gemm(xPbf, res_t, rootP, rootP, NP, DIN);             // += residual proj

  // ---- layer 1 aggregation + epilogue
  zero(aggD, (long long)ND * DHID);
  zero(aggP, (long long)NP * DHID);
  scatter(msgP, src_use, dst_use, aggD);
  scatter(msgD, src_rev, dst_rev, aggP);
  k_epilogue<<<ND, 256, 0, stream>>>(aggD, cntD, rootD, nullptr, b1_use, res_b,
                                     ln1_g, ln1_b, hD, hDbf);
  k_epilogue<<<NP, 256, 0, stream>>>(aggP, cntP, rootP, nullptr, b1_rev, res_b,
                                     ln1_g, ln1_b, hP, hPbf);

  // ---- layer 2 GEMMs
  gemm(hPbf, w2u_t, nullptr, msgP, NP, DHID);
  gemm(hDbf, w2r_t, nullptr, msgD, ND, DHID);
  gemm(hDbf, r2u_t, nullptr, rootD, ND, DHID);
  gemm(hPbf, r2r_t, nullptr, rootP, NP, DHID);

  // ---- layer 2 aggregation + epilogue (writes final outputs)
  zero(aggD, (long long)ND * DHID);
  zero(aggP, (long long)NP * DHID);
  scatter(msgP, src_use, dst_use, aggD);
  scatter(msgD, src_rev, dst_rev, aggP);
  k_epilogue<<<ND, 256, 0, stream>>>(aggD, cntD, rootD, hD, b2_use, nullptr,
                                     ln2_g, ln2_b, y_ds, nullptr);
  k_epilogue<<<NP, 256, 0, stream>>>(aggP, cntP, rootP, hP, b2_rev, nullptr,
                                     ln2_g, ln2_b, y_pp, nullptr);
}
